// LSTM_27900107555442
// MI455X (gfx1250) — matrix operand order repaired and checked by reading, not compiled, not run on hardware
//
#include <hip/hip_runtime.h>

typedef __attribute__((ext_vector_type(16))) __bf16 v16bf;
typedef __attribute__((ext_vector_type(8)))  __bf16 v8bf;
typedef __attribute__((ext_vector_type(8)))  float  v8f;

#define B_  2048
#define T_  512
#define I_  6
#define H_  50
#define O_  3

// f32 -> bf16 round-to-nearest-even (bit trick; NaN case irrelevant here)
static __device__ __forceinline__ unsigned short f2bf(float f) {
  union { float f; unsigned u; } x; x.f = f;
  unsigned r = x.u + 0x7fffu + ((x.u >> 16) & 1u);
  return (unsigned short)(r >> 16);
}

#if __has_builtin(__builtin_amdgcn_tanhf)
// CDNA5 native v_tanh_f32: 1 trans op per activation.
static __device__ __forceinline__ float ftanh(float x) {
  return __builtin_amdgcn_tanhf(x);
}
static __device__ __forceinline__ float fsig(float x) {
  // sigmoid(x) = 0.5*tanh(0.5x) + 0.5
  return __builtin_fmaf(0.5f, __builtin_amdgcn_tanhf(0.5f * x), 0.5f);
}
#else
static __device__ __forceinline__ float fsig(float x) {
  float e = __builtin_amdgcn_exp2f(-1.44269504f * x);   // v_exp_f32
  return __builtin_amdgcn_rcpf(1.0f + e);               // v_rcp_f32
}
static __device__ __forceinline__ float ftanh(float x) {
  return 2.0f * fsig(2.0f * x) - 1.0f;
}
#endif

// A-matrix fragment (16x32 bf16): lane<16 holds K base..base+7 then base+16..+23
// (base=0); lane>=16 holds K 8..15 then 24..31.  Two 16B LDS loads per chunk.
static __device__ __forceinline__ v16bf ldA(const unsigned short* p) {
  v8bf lo = *reinterpret_cast<const v8bf*>(p);        // K +0..7   (16B)
  v8bf hi = *reinterpret_cast<const v8bf*>(p + 16);   // K +16..23 (16B, +32B)
  return __builtin_shufflevector(lo, hi, 0,1,2,3,4,5,6,7,8,9,10,11,12,13,14,15);
}
// B-matrix fragment (32x16 bf16): lane holds one column n, K base..base+7 then base+16..+23
// (lanes<16: K 0..7,16..23, lanes>=16: K 8..15,24..31), column-major.
static __device__ __forceinline__ v16bf ldB(const unsigned short* p) {
  v8bf lo = *reinterpret_cast<const v8bf*>(p);
  v8bf hi = *reinterpret_cast<const v8bf*>(p + 16);
  return __builtin_shufflevector(lo, hi, 0,1,2,3,4,5,6,7,8,9,10,11,12,13,14,15);
}

__global__ __launch_bounds__(128) void lstm_fused_wmma(
    const float* __restrict__ x,    const float* __restrict__ Wih,
    const float* __restrict__ Whh,  const float* __restrict__ bih,
    const float* __restrict__ bhh,  const float* __restrict__ Wout,
    const float* __restrict__ bout, float* __restrict__ out)
{
  // Combined weights, column-major bf16: Wls[col][k], col = gate*64 + jj (jj padded 50->64),
  // k: 0..5 = W_ih row, 6..55 = W_hh row, 56..63 = 0.
  __shared__ __align__(16) unsigned short Wls[4 * 64 * 64];    // 32 KB
  __shared__ __align__(16) unsigned short Als[2][16 * 64];     // A ping-pong, [m][k] bf16, 4 KB
  __shared__ float Hls[16 * 64];                               // final h staging, 4 KB

  const int tid  = threadIdx.x;
  const int wave = tid >> 5;          // hidden sub-tile 0..3 (16 hidden units each)
  const int lane = tid & 31;
  const int n    = lane & 15;         // N index within tile (hidden unit)
  const int hi   = lane >> 4;         // lane half
  const int b0   = blockIdx.x * 16;   // batch tile base

  // ---- one-time: stage combined padded weights as bf16 into LDS ----
  for (int idx = tid; idx < 4 * 64 * 64; idx += 128) {
    int col = idx >> 6, k = idx & 63;
    int g = col >> 6, jj = col & 63;
    float v = 0.0f;
    if (jj < H_) {
      int row = g * H_ + jj;                       // PyTorch gate order i,f,g,o
      if (k < I_)           v = Wih[row * I_ + k];
      else if (k < I_ + H_) v = Whh[row * H_ + (k - I_)];
    }
    Wls[idx] = f2bf(v);
  }
  // zero both A buffers (h0 = 0, and K=56..63 zero padding stays zero forever)
  for (int idx = tid; idx < 2 * 16 * 64; idx += 128)
    (&Als[0][0])[idx] = 0;
  __syncthreads();

  // ---- preload this wave's 8 B fragments (4 gates x 2 K-chunks) into VGPRs ----
  v16bf Bf[4][2];
#pragma unroll
  for (int g = 0; g < 4; ++g)
#pragma unroll
    for (int c = 0; c < 2; ++c)
      Bf[g][c] = ldB(&Wls[(g * 64 + wave * 16 + n) * 64 + c * 32 + hi * 8]);

  const int j = wave * 16 + n;        // global hidden index owned by this lane
  float bias[4];
#pragma unroll
  for (int g = 0; g < 4; ++g)
    bias[g] = (j < H_) ? (bih[g * H_ + j] + bhh[g * H_ + j]) : 0.0f;

  // wave 0 writes x_0 into A[0]: lanes 0-15 -> elems 0..2, lanes 16-31 -> elems 3..5
  if (wave == 0) {
    const float* xp = x + ((size_t)(b0 + n) * T_) * I_ + hi * 3;
    Als[0][n * 64 + hi * 3 + 0] = f2bf(xp[0]);
    Als[0][n * 64 + hi * 3 + 1] = f2bf(xp[1]);
    Als[0][n * 64 + hi * 3 + 2] = f2bf(xp[2]);
  }
  __syncthreads();

  float c8[8], h8[8];
#pragma unroll
  for (int e = 0; e < 8; ++e) { c8[e] = 0.0f; h8[e] = 0.0f; }

  // ================= recurrence =================
  for (int t = 0; t < T_; ++t) {
    const int cur = t & 1, nxt = cur ^ 1;
    const unsigned short* Ab = &Als[cur][0];
    // A fragments: [16 batch x K], chunk 0 = K 0..31, chunk 1 = K 32..63
    v16bf a0 = ldA(&Ab[n * 64 +  0 + hi * 8]);
    v16bf a1 = ldA(&Ab[n * 64 + 32 + hi * 8]);

    v8f z = {};
    v8f d[4];
#pragma unroll
    for (int g = 0; g < 4; ++g) {
      v8f acc = __builtin_amdgcn_wmma_f32_16x16x32_bf16(
          false, a0, false, Bf[g][0], (short)0, z, false, false);
      d[g] = __builtin_amdgcn_wmma_f32_16x16x32_bf16(
          false, a1, false, Bf[g][1], (short)0, acc, false, false);
    }

    // cell update: accumulator element e -> batch m = e + 8*hi, hidden j
    unsigned short* An = &Als[nxt][0];
#pragma unroll
    for (int e = 0; e < 8; ++e) {
      float gi = d[0][e] + bias[0];
      float gf = d[1][e] + bias[1];
      float gg = d[2][e] + bias[2];
      float go = d[3][e] + bias[3];
      float cc = fsig(gf) * c8[e] + fsig(gi) * ftanh(gg);
      float hh = fsig(go) * ftanh(cc);
      c8[e] = cc;
      h8[e] = hh;
    }
    if (j < H_) {
#pragma unroll
      for (int e = 0; e < 8; ++e)
        An[(e + 8 * hi) * 64 + I_ + j] = f2bf(h8[e]);
    }
    if (wave == 0 && t + 1 < T_) {
      const float* xp = x + ((size_t)(b0 + n) * T_ + (t + 1)) * I_ + hi * 3;
      An[n * 64 + hi * 3 + 0] = f2bf(xp[0]);
      An[n * 64 + hi * 3 + 1] = f2bf(xp[1]);
      An[n * 64 + hi * 3 + 2] = f2bf(xp[2]);
    }
    __syncthreads();
  }

  // ================= head: Linear(50,3) + softmax =================
  if (j < H_) {
#pragma unroll
    for (int e = 0; e < 8; ++e)
      Hls[(e + 8 * hi) * 64 + j] = h8[e];
  }
  __syncthreads();

  if (wave == 0 && lane < 16) {
    const int m = lane;
    float a0 = bout[0], a1 = bout[1], a2 = bout[2];
    for (int jj = 0; jj < H_; ++jj) {
      float hv = Hls[m * 64 + jj];
      a0 += hv * Wout[0 * H_ + jj];
      a1 += hv * Wout[1 * H_ + jj];
      a2 += hv * Wout[2 * H_ + jj];
    }
    float mx = fmaxf(a0, fmaxf(a1, a2));
    float e0 = __builtin_amdgcn_exp2f((a0 - mx) * 1.44269504f);
    float e1 = __builtin_amdgcn_exp2f((a1 - mx) * 1.44269504f);
    float e2 = __builtin_amdgcn_exp2f((a2 - mx) * 1.44269504f);
    float rs = __builtin_amdgcn_rcpf(e0 + e1 + e2);
    float* op = out + (size_t)(b0 + m) * O_;
    op[0] = e0 * rs;
    op[1] = e1 * rs;
    op[2] = e2 * rs;
  }
}

extern "C" void kernel_launch(void* const* d_in, const int* in_sizes, int n_in,
                              void* d_out, int out_size, void* d_ws, size_t ws_size,
                              hipStream_t stream) {
  (void)in_sizes; (void)n_in; (void)out_size; (void)d_ws; (void)ws_size;
  const float* x    = (const float*)d_in[0];
  const float* Wih  = (const float*)d_in[1];
  const float* Whh  = (const float*)d_in[2];
  const float* bih  = (const float*)d_in[3];
  const float* bhh  = (const float*)d_in[4];
  const float* Wout = (const float*)d_in[5];
  const float* bout = (const float*)d_in[6];
  lstm_fused_wmma<<<dim3(B_ / 16), dim3(128), 0, stream>>>(
      x, Wih, Whh, bih, bhh, Wout, bout, (float*)d_out);
}
